// StandardMamba_26164940767414
// MI455X (gfx1250) — matrix operand order repaired and checked by reading, not compiled, not run on hardware
//
#include <hip/hip_runtime.h>
#include <hip/hip_bf16.h>
#include <math.h>

typedef __attribute__((ext_vector_type(2))) float v2f;
typedef __attribute__((ext_vector_type(8))) float v8f;
typedef __attribute__((ext_vector_type(8))) __bf16 v8bf;
typedef __attribute__((ext_vector_type(16))) __bf16 v16bf;

#define D_MODEL 1024
#define N_LAYERS 4
#define D_STATE 16
#define D_CONV 4
#define D_INNER 2048
#define DT_RANK 64
#define SEQ 2048
#define BATCH 2
#define NTOK (BATCH * SEQ) /* 4096 tokens */
#define LN_EPS 1e-5f

#define BM 64
#define BN 64
#define BK 32

// bf16 tile pitch: 40 elems = 80B = 20 DWORDs -> 16B fragment reads tile all
// 64 LDS banks conflict-free (start banks n*20 mod 64 stride the bank space).
#define BKH 32
#define HPITCH 40

__device__ __forceinline__ float silu_f(float x) { return x / (1.f + expf(-x)); }
__device__ __forceinline__ float softplus_f(float x) {
  return (x > 20.f) ? x : log1pf(expf(x));
}

// CDNA5 async global->LDS copy (VGLOBAL GV mode, tracked by ASYNCcnt).
// vdst = per-lane LDS byte offset, vaddr = per-lane 64-bit global address.
__device__ __forceinline__ void async_ld_b128(unsigned lds_off,
                                              const void* gaddr) {
  asm volatile("global_load_async_to_lds_b128 %0, %1, off"
               :
               : "v"(lds_off), "v"(gaddr)
               : "memory");
}
__device__ __forceinline__ void wait_asynccnt0() {
  asm volatile("s_wait_asynccnt 0x0" ::: "memory");
}

// ---------------------------------------------------------------------------
// LayerNorm over last dim (D_MODEL), bf16 output (feeds bf16 WMMA in_proj).
// ---------------------------------------------------------------------------
__global__ __launch_bounds__(256) void ln_kernel(const float* __restrict__ x,
                                                 const float* __restrict__ w,
                                                 const float* __restrict__ b,
                                                 __bf16* __restrict__ out) {
  __shared__ float red[256];
  const int row = blockIdx.x;
  const int tid = threadIdx.x;
  const float* xr = x + (size_t)row * D_MODEL;

  float s = 0.f;
  for (int i = tid; i < D_MODEL; i += 256) s += xr[i];
  red[tid] = s;
  __syncthreads();
  for (int off = 128; off > 0; off >>= 1) {
    if (tid < off) red[tid] += red[tid + off];
    __syncthreads();
  }
  const float mu = red[0] * (1.f / D_MODEL);
  __syncthreads();

  float v = 0.f;
  for (int i = tid; i < D_MODEL; i += 256) {
    float d = xr[i] - mu;
    v += d * d;
  }
  red[tid] = v;
  __syncthreads();
  for (int off = 128; off > 0; off >>= 1) {
    if (tid < off) red[tid] += red[tid + off];
    __syncthreads();
  }
  const float rstd = rsqrtf(red[0] * (1.f / D_MODEL) + LN_EPS);

  __bf16* orow = out + (size_t)row * D_MODEL;
  for (int i = tid; i < D_MODEL; i += 256)
    orow[i] = (__bf16)((xr[i] - mu) * rstd * w[i] + b[i]);
}

// ---------------------------------------------------------------------------
// fp32 -> bf16 conversion (weights; per-call, deterministic).
// ---------------------------------------------------------------------------
__global__ __launch_bounds__(256) void cvt_f32_bf16(const float* __restrict__ in,
                                                    __bf16* __restrict__ out,
                                                    long n) {
  const long i = (long)blockIdx.x * 256 + threadIdx.x;
  if (i < n) out[i] = (__bf16)in[i];
}

// ---------------------------------------------------------------------------
// bf16 WMMA GEMM: C[M,N] = A[M,K] * W[N,K]^T, fp32 accumulate.
//   mode 0: plain      mode 2: acc + res[m,n]
// Requires M%64==0, N%64==0, K%32==0 (true for in_proj / out_proj).
// Block = 256 threads (8 wave32), 64x64 tile; wave owns 16x32 via two
// v_wmma_f32_16x16x32_bf16 per K-stage. LDS tiles stay row-major (fragments
// are contiguous 16B K-runs per lane, k = 16*half + s order, same for A and B) and are
// staged with double-buffered GLOBAL_LOAD_ASYNC_TO_LDS_B128 (ASYNCcnt), so
// next-stage copies overlap current-stage WMMAs with ONE barrier per stage.
// ---------------------------------------------------------------------------
__global__ __launch_bounds__(256) void gemm_bf16_wmma(
    const __bf16* __restrict__ A, const __bf16* __restrict__ W,
    float* __restrict__ C, const float* __restrict__ res, int M, int N, int K,
    int mode) {
  __shared__ __bf16 As[2][BM][HPITCH];
  __shared__ __bf16 Ws[2][BN][HPITCH];

  const int bm = blockIdx.y * BM;
  const int bn = blockIdx.x * BN;
  const int tid = threadIdx.x;
  const int wave = tid >> 5;
  const int lane = tid & 31;
  const int wm = (wave >> 1) * 16;  // 0,16,32,48
  const int wn = (wave & 1) * 32;   // 0,32

  v8f acc0 = {};
  v8f acc1 = {};

  const int lr = tid >> 2;       // 0..63: tile row
  const int lc = (tid & 3) * 8;  // 0,8,16,24: 8 bf16 (16B) per thread

  // Per-lane fragment geometry (contiguous K order k = 16*half + s, same for A and B):
  const int half = lane >> 4;  // lane half selects K sub-block
  const int fr = lane & 15;    // M (A) or N (B) index
  const int akb = half * 16;   // A: elems 0..15 = K akb..akb+15
  const int bkb = half * 16;   // B: elems 0..15 = K bkb..bkb+15

  // Per-lane LDS byte offsets for async staging (low 32 bits of generic
  // shared-aperture address = LDS offset).
  const unsigned asOff[2] = {(unsigned)(size_t)&As[0][lr][lc],
                             (unsigned)(size_t)&As[1][lr][lc]};
  const unsigned wsOff[2] = {(unsigned)(size_t)&Ws[0][lr][lc],
                             (unsigned)(size_t)&Ws[1][lr][lc]};
  const __bf16* aptr = &A[(size_t)(bm + lr) * K + lc];
  const __bf16* wptr = &W[(size_t)(bn + lr) * K + lc];

  // Prologue: stage K-tile 0 into buffer 0.
  async_ld_b128(asOff[0], aptr);
  async_ld_b128(wsOff[0], wptr);

  for (int k0 = 0; k0 < K; k0 += BKH) {
    const int cb = (k0 / BKH) & 1;
    wait_asynccnt0();  // my async copies for buffer cb have landed
    __syncthreads();   // everyone's landed; buffer cb^1 no longer being read

    if (k0 + BKH < K) {  // overlap next-stage copies with this stage's WMMAs
      async_ld_b128(asOff[cb ^ 1], aptr + (k0 + BKH));
      async_ld_b128(wsOff[cb ^ 1], wptr + (k0 + BKH));
    }
    if (k0 + 2 * BKH < K) {  // CDNA5 global_prefetch_b8 two stages ahead
      __builtin_prefetch(aptr + (k0 + 2 * BKH), 0, 0);
      __builtin_prefetch(wptr + (k0 + 2 * BKH), 0, 0);
    }

    v8bf alo = *(const v8bf*)&As[cb][wm + fr][akb];
    v8bf ahi = *(const v8bf*)&As[cb][wm + fr][akb + 8];
    v16bf a = __builtin_shufflevector(alo, ahi, 0, 1, 2, 3, 4, 5, 6, 7, 8, 9,
                                      10, 11, 12, 13, 14, 15);

    v8bf b0lo = *(const v8bf*)&Ws[cb][wn + fr][bkb];
    v8bf b0hi = *(const v8bf*)&Ws[cb][wn + fr][bkb + 8];
    v16bf b0 = __builtin_shufflevector(b0lo, b0hi, 0, 1, 2, 3, 4, 5, 6, 7, 8,
                                       9, 10, 11, 12, 13, 14, 15);

    v8bf b1lo = *(const v8bf*)&Ws[cb][wn + 16 + fr][bkb];
    v8bf b1hi = *(const v8bf*)&Ws[cb][wn + 16 + fr][bkb + 8];
    v16bf b1 = __builtin_shufflevector(b1lo, b1hi, 0, 1, 2, 3, 4, 5, 6, 7, 8,
                                       9, 10, 11, 12, 13, 14, 15);

    acc0 = __builtin_amdgcn_wmma_f32_16x16x32_bf16(false, a, false, b0,
                                                   (short)0, acc0, false,
                                                   false);
    acc1 = __builtin_amdgcn_wmma_f32_16x16x32_bf16(false, a, false, b1,
                                                   (short)0, acc1, false,
                                                   false);
  }

  // C/D layout: VGPR r = rows r (lanes 0-15) / r+8 (lanes 16-31); N = lane%16.
  const int mbase = bm + wm + (half << 3);
  const int gn0 = bn + wn + fr;
  const int gn1 = gn0 + 16;
#pragma unroll
  for (int r = 0; r < 8; ++r) {
    const size_t rowoff = (size_t)(mbase + r) * N;
    float v0 = acc0[r];
    float v1 = acc1[r];
    if (mode == 2) {
      v0 += res[rowoff + gn0];
      v1 += res[rowoff + gn1];
    }
    C[rowoff + gn0] = v0;
    C[rowoff + gn1] = v1;
  }
}

// ---------------------------------------------------------------------------
// Exact-fp32 WMMA GEMM (V_WMMA_F32_16X16X4_F32) for the precision-sensitive
// SSM-parameter projections (x_proj -> B/C, dt projection -> exp path).
//   mode 0: plain    mode 1: softplus(acc + bias[n])
// ---------------------------------------------------------------------------
__global__ __launch_bounds__(256) void gemm_f32_wmma(
    const float* __restrict__ A, const float* __restrict__ W,
    float* __restrict__ C, const float* __restrict__ bias, int M, int N, int K,
    int lda, int mode) {
  __shared__ float As[BM][BK + 1];
  __shared__ float Bs[BK][BN + 1];

  const int bm = blockIdx.y * BM;
  const int bn = blockIdx.x * BN;
  const int tid = threadIdx.x;
  const int wave = tid >> 5;
  const int lane = tid & 31;
  const int wm = (wave >> 1) * 16;
  const int wn = (wave & 1) * 32;

  v8f acc0 = {};
  v8f acc1 = {};

  for (int k0 = 0; k0 < K; k0 += BK) {
    for (int i = tid; i < BM * BK; i += 256) {
      const int r = i >> 5;
      const int c = i & 31;
      const int gm = bm + r;
      As[r][c] = (gm < M) ? A[(size_t)gm * lda + (k0 + c)] : 0.f;
    }
    for (int i = tid; i < BK * BN; i += 256) {
      const int n = i >> 5;
      const int kk = i & 31;
      const int gn = bn + n;
      Bs[kk][n] = (gn < N) ? W[(size_t)gn * K + (k0 + kk)] : 0.f;
    }
    __syncthreads();

    const int khalf = (lane >> 4) << 1;
    const int mrow = lane & 15;
    const int ncol = lane & 15;
    for (int kk = 0; kk < BK; kk += 4) {
      v2f a;
      a.x = As[wm + mrow][kk + khalf];
      a.y = As[wm + mrow][kk + khalf + 1];
      v2f b0, b1;
      b0.x = Bs[kk + khalf][wn + ncol];
      b0.y = Bs[kk + khalf + 1][wn + ncol];
      b1.x = Bs[kk + khalf][wn + 16 + ncol];
      b1.y = Bs[kk + khalf + 1][wn + 16 + ncol];
      acc0 = __builtin_amdgcn_wmma_f32_16x16x4_f32(false, a, false, b0,
                                                   (short)0, acc0, false,
                                                   false);
      acc1 = __builtin_amdgcn_wmma_f32_16x16x4_f32(false, a, false, b1,
                                                   (short)0, acc1, false,
                                                   false);
    }
    __syncthreads();
  }

  const int mbase = bm + wm + ((lane >> 4) << 3);
  const int gn0 = bn + wn + (lane & 15);
  const int gn1 = gn0 + 16;
  for (int r = 0; r < 8; ++r) {
    const int gm = mbase + r;
    if (gm >= M) continue;
    const size_t rowoff = (size_t)gm * N;
    float v0 = acc0[r];
    float v1 = acc1[r];
    if (mode == 1) {
      if (gn0 < N) v0 = softplus_f(v0 + bias[gn0]);
      if (gn1 < N) v1 = softplus_f(v1 + bias[gn1]);
    }
    if (gn0 < N) C[rowoff + gn0] = v0;
    if (gn1 < N) C[rowoff + gn1] = v1;
  }
}

// ---------------------------------------------------------------------------
// Depthwise causal conv (width 4) + bias + SiLU.  xz holds [x_in | z] (4096).
// ---------------------------------------------------------------------------
__global__ __launch_bounds__(256) void conv_silu_kernel(
    const float* __restrict__ xz, const float* __restrict__ cw,
    const float* __restrict__ cb, float* __restrict__ xc) {
  const long idx = (long)blockIdx.x * 256 + threadIdx.x;
  if (idx >= (long)NTOK * D_INNER) return;
  const int d = (int)(idx % D_INNER);
  const int t = (int)(idx / D_INNER);  // b*SEQ + s
  const int s = t % SEQ;
  float acc = cb[d];
#pragma unroll
  for (int j = 0; j < D_CONV; ++j) {
    const int sp = s - (D_CONV - 1) + j;
    if (sp >= 0)
      acc += cw[d * D_CONV + j] * xz[(size_t)(t - (s - sp)) * (2 * D_INNER) + d];
  }
  xc[(size_t)t * D_INNER + d] = silu_f(acc);
}

// ---------------------------------------------------------------------------
// Selective scan; fused D-skip + SiLU(z) gate; bf16 output (feeds out_proj).
// Block = 256 threads = 16 channels x 16 states; wave-local shuffle reduce.
// ---------------------------------------------------------------------------
__global__ __launch_bounds__(256) void scan_kernel(
    const float* __restrict__ dt, const float* __restrict__ xz,
    const float* __restrict__ xc, const float* __restrict__ dbl,
    const float* __restrict__ A_log, const float* __restrict__ Dp,
    __bf16* __restrict__ y) {
  const int b = blockIdx.x / (D_INNER / 16);
  const int dg = blockIdx.x % (D_INNER / 16);
  const int dloc = threadIdx.x >> 4;
  const int n = threadIdx.x & 15;
  const int d = dg * 16 + dloc;

  const float A = -expf(A_log[d * D_STATE + n]);
  const float Dskip = Dp[d];
  float h = 0.f;

  __shared__ float sBC[32];
  const size_t tbase = (size_t)b * SEQ;

  for (int s = 0; s < SEQ; ++s) {
    const size_t t = tbase + s;
    if (threadIdx.x < 32)
      sBC[threadIdx.x] = dbl[t * 96 + DT_RANK + threadIdx.x];
    __syncthreads();

    const float dtv = dt[t * D_INNER + d];
    const float u = xc[t * D_INNER + d];
    const float Bt = sBC[n];
    const float Ct = sBC[16 + n];

    h = h * expf(dtv * A) + (dtv * u) * Bt;
    float yv = h * Ct;
    yv += __shfl_xor(yv, 1, 32);
    yv += __shfl_xor(yv, 2, 32);
    yv += __shfl_xor(yv, 4, 32);
    yv += __shfl_xor(yv, 8, 32);

    if (n == 0) {
      const float z = xz[t * (2 * D_INNER) + D_INNER + d];
      y[t * D_INNER + d] = (__bf16)((yv + Dskip * u) * silu_f(z));
    }
    __syncthreads();
  }
}

// ---------------------------------------------------------------------------
extern "C" void kernel_launch(void* const* d_in, const int* in_sizes, int n_in,
                              void* d_out, int out_size, void* d_ws,
                              size_t ws_size, hipStream_t stream) {
  const float* x0 = (const float*)d_in[0];
  const float* in_proj_w = (const float*)d_in[1];
  const float* conv_w = (const float*)d_in[2];
  const float* conv_b = (const float*)d_in[3];
  const float* x_proj_w = (const float*)d_in[4];
  const float* dt_proj_w = (const float*)d_in[5];
  const float* dt_proj_b = (const float*)d_in[6];
  const float* A_log = (const float*)d_in[7];
  const float* D_skip = (const float*)d_in[8];
  const float* out_proj_w = (const float*)d_in[9];
  const float* ln_w = (const float*)d_in[10];
  const float* ln_b = (const float*)d_in[11];

  float* ws = (float*)d_ws;
  float* cur = ws;                              // NTOK x D_MODEL f32
  float* xz = cur + (size_t)NTOK * D_MODEL;     // NTOK x 2*D_INNER f32
  float* xc = xz + (size_t)NTOK * 2 * D_INNER;  // NTOK x D_INNER  f32
  float* dbl = xc + (size_t)NTOK * D_INNER;     // NTOK x 96       f32
  float* dtb = dbl + (size_t)NTOK * 96;         // NTOK x D_INNER  f32
  float* endf = dtb + (size_t)NTOK * D_INNER;
  __bf16* xn_bf = (__bf16*)endf;                          // NTOK x D_MODEL
  __bf16* y_bf = xn_bf + (size_t)NTOK * D_MODEL;          // NTOK x D_INNER
  __bf16* wi_bf = y_bf + (size_t)NTOK * D_INNER;          // 2*D_INNER x D_MODEL
  __bf16* wo_bf = wi_bf + (size_t)2 * D_INNER * D_MODEL;  // D_MODEL x D_INNER

  hipMemcpyAsync(cur, x0, (size_t)NTOK * D_MODEL * sizeof(float),
                 hipMemcpyDeviceToDevice, stream);

  for (int l = 0; l < N_LAYERS; ++l) {
    ln_kernel<<<NTOK, 256, 0, stream>>>(cur, ln_w + (size_t)l * D_MODEL,
                                        ln_b + (size_t)l * D_MODEL, xn_bf);

    // in_proj (bf16 WMMA): (4096 x 1024) * (4096 x 1024)^T -> 4096 x 4096
    {
      const long nw = (long)2 * D_INNER * D_MODEL;
      cvt_f32_bf16<<<(nw + 255) / 256, 256, 0, stream>>>(
          in_proj_w + (size_t)l * nw, wi_bf, nw);
      dim3 g((2 * D_INNER) / BN, NTOK / BM);
      gemm_bf16_wmma<<<g, 256, 0, stream>>>(xn_bf, wi_bf, xz, nullptr, NTOK,
                                            2 * D_INNER, D_MODEL, 0);
    }

    conv_silu_kernel<<<(NTOK * D_INNER + 255) / 256, 256, 0, stream>>>(
        xz, conv_w + (size_t)l * D_INNER * D_CONV, conv_b + (size_t)l * D_INNER,
        xc);

    // x_proj (exact fp32 WMMA): (4096 x 2048) * (96 x 2048)^T -> 4096 x 96
    {
      dim3 g((96 + BN - 1) / BN, NTOK / BM);
      gemm_f32_wmma<<<g, 256, 0, stream>>>(
          xc, x_proj_w + (size_t)l * 96 * D_INNER, dbl, nullptr, NTOK, 96,
          D_INNER, D_INNER, 0);
    }

    // dt (exact fp32 WMMA + softplus/bias): dtr = dbl[:, :64] (lda = 96)
    {
      dim3 g(D_INNER / BN, NTOK / BM);
      gemm_f32_wmma<<<g, 256, 0, stream>>>(
          dbl, dt_proj_w + (size_t)l * D_INNER * DT_RANK, dtb,
          dt_proj_b + (size_t)l * D_INNER, NTOK, D_INNER, DT_RANK, 96, 1);
    }

    scan_kernel<<<BATCH * (D_INNER / 16), 256, 0, stream>>>(
        dtb, xz, xc, dbl, A_log + (size_t)l * D_INNER * D_STATE,
        D_skip + (size_t)l * D_INNER, y_bf);

    // out_proj (bf16 WMMA) + residual: (4096 x 2048)*(1024 x 2048)^T + cur
    {
      const long nw = (long)D_MODEL * D_INNER;
      cvt_f32_bf16<<<(nw + 255) / 256, 256, 0, stream>>>(
          out_proj_w + (size_t)l * nw, wo_bf, nw);
      float* dst = (l == N_LAYERS - 1) ? (float*)d_out : cur;
      dim3 g(D_MODEL / BN, NTOK / BM);
      gemm_bf16_wmma<<<g, 256, 0, stream>>>(y_bf, wo_bf, dst, cur, NTOK,
                                            D_MODEL, D_INNER, 2);
    }
  }
}
